// AnnsHNSW_42331197670181
// MI455X (gfx1250) — matrix operand order repaired and checked by reading, not compiled, not run on hardware
//
#include <hip/hip_runtime.h>
#include <hip/hip_bf16.h>

// Problem constants (match reference)
#define BH_TOT 32      // B*H = 2*16
#define NQ     2048
#define NK     2048
#define DIMS   64
#define KSEL   64      // sample_size
#define QT     16      // queries per block (kernel 1)

typedef __bf16 v16bf __attribute__((ext_vector_type(16)));
typedef float  v8f   __attribute__((ext_vector_type(8)));

// ---------------------------------------------------------------------------
// Kernel 1: bf16 WMMA score panel (16 queries x 2048 keys in LDS) + top-64
// 512 threads = 16 wave32 waves. Dynamic LDS: 16*2048*4 + 16*64*4 bytes.
// ---------------------------------------------------------------------------
__global__ __launch_bounds__(512)
void qnf_score_topk_kernel(const float* __restrict__ query,
                           const float* __restrict__ key,
                           int* __restrict__ sampled /* [BH][NQ][KSEL] */)
{
    extern __shared__ char smem_raw[];
    float* scores = (float*)smem_raw;                          // [QT][NK]
    int*   topk   = (int*)(smem_raw + (size_t)QT * NK * 4);    // [QT][KSEL]

    const int bh   = blockIdx.x / (NQ / QT);
    const int q0   = (blockIdx.x % (NQ / QT)) * QT;
    const int tid  = threadIdx.x;
    const int wave = tid >> 5;
    const int lane = tid & 31;
    const int p    = lane >> 4;   // half-wave select
    const int ln   = lane & 15;

    // ---- A fragments: 16-bit A matrix 16x32, contiguous k order, same as B ----
    // lane l (row M = l&15), elem e -> K = 16*(l>>4) + e
    const float* qrow = query + ((size_t)bh * NQ + q0 + ln) * DIMS;
    v16bf a0, a1;
#pragma unroll
    for (int e = 0; e < 16; ++e) {
        int kd = (p << 4) + e;
        a0[e] = (__bf16)qrow[kd];
        a1[e] = (__bf16)qrow[kd + 32];
    }

    // ---- 8 contiguous key tiles of 16 per wave: 128 keys each ----
    const int tile0 = wave * 8;
    for (int t = 0; t < 8; ++t) {
        const int kbase = (tile0 + t) * 16;
        const float* krow = key + ((size_t)bh * NK + kbase + ln) * DIMS;
        // speculative prefetch of next tile's row -> global_prefetch_b8
        __builtin_prefetch(krow + 16 * DIMS, 0, 1);

        // B fragment 32x16: lane l holds column N = l&15; elem e -> K = 16*p + e
        v16bf b0, b1;
#pragma unroll
        for (int e = 0; e < 16; ++e) {
            int kd = (p << 4) + e;
            b0[e] = (__bf16)krow[kd];
            b1[e] = (__bf16)krow[kd + 32];
        }

        v8f c = {};
        c = __builtin_amdgcn_wmma_f32_16x16x32_bf16(false, a0, false, b0,
                                                    (short)0, c, false, false);
        c = __builtin_amdgcn_wmma_f32_16x16x32_bf16(false, a1, false, b1,
                                                    (short)0, c, false, false);

        // C layout: vgpr v, lane l -> M = v + 8*(l>>4), N = l&15
#pragma unroll
        for (int v = 0; v < 8; ++v) {
            int m = v + (p << 3);
            scores[m * NK + kbase + ln] = c[v];
        }
    }
    __syncthreads();

    // ---- top-64 per query: wave w owns query w; iterative argmax+mask ----
    float* srow = scores + wave * NK;
    for (int it = 0; it < KSEL; ++it) {
        float bv = -__builtin_inff();
        int   bi = NK;
        for (int j = lane; j < NK; j += 32) {
            float v = srow[j];
            if (v > bv || (v == bv && j < bi)) { bv = v; bi = j; }
        }
#pragma unroll
        for (int off = 16; off > 0; off >>= 1) {
            float ov = __shfl_xor(bv, off, 32);
            int   oi = __shfl_xor(bi, off, 32);
            if (ov > bv || (ov == bv && oi < bi)) { bv = ov; bi = oi; }
        }
        if (lane == 0) {
            topk[wave * KSEL + it] = bi;
            srow[bi] = -__builtin_inff();   // mask out winner
        }
    }
    __syncthreads();

    // ---- emit sampled_set (descending score == ascending d2, stable ties) ----
    for (int t = lane; t < KSEL; t += 32) {
        sampled[(((size_t)bh * NQ) + q0 + wave) * KSEL + t] = topk[wave * KSEL + t];
    }
}

// ---------------------------------------------------------------------------
// Kernel 2: per-(b,h) stable argsort of labels (rank-count) + strided gather
// ---------------------------------------------------------------------------
__global__ __launch_bounds__(1024)
void sort_pick_kernel(const int* __restrict__ sampled,
                      int* __restrict__ qsort_out,  // [BH][NQ]
                      int* __restrict__ pick_out)   // [BH][NQ]
{
    __shared__ int label[NQ];
    __shared__ int sortidx[NQ];
    const int bh  = blockIdx.x;
    const int tid = threadIdx.x;

    for (int q = tid; q < NQ; q += 1024)
        label[q] = sampled[(((size_t)bh * NQ) + q) * KSEL];   // nearest neighbor
    __syncthreads();

    // stable rank: #(label[j] < label[q]) + #(label[j] == label[q] && j < q)
    for (int q = tid; q < NQ; q += 1024) {
        const int lq = label[q];
        int r = 0;
        for (int j = 0; j < NQ; ++j) {
            int lj = label[j];
            r += (lj < lq) || (lj == lq && j < q);
        }
        sortidx[r] = q;
        qsort_out[(size_t)bh * NQ + r] = q;   // argsort output
    }
    __syncthreads();

    // picked rows = sorted positions 0,64,...,1984; flattened gives exactly NQ ints
    for (int i = tid; i < NQ; i += 1024) {
        int row = i >> 6;
        int src = sortidx[row << 6];
        pick_out[(size_t)bh * NQ + i] =
            sampled[(((size_t)bh * NQ) + src) * KSEL + (i & 63)];
    }
}

// ---------------------------------------------------------------------------
extern "C" void kernel_launch(void* const* d_in, const int* in_sizes, int n_in,
                              void* d_out, int out_size, void* d_ws, size_t ws_size,
                              hipStream_t stream)
{
    (void)in_sizes; (void)n_in; (void)out_size; (void)ws_size;
    const float* query = (const float*)d_in[0];
    const float* key   = (const float*)d_in[1];
    // d_in[2] = sample_size (==64), baked in as KSEL.

    int* sampled   = (int*)d_ws;                 // [BH][NQ][KSEL] = 16.8 MB
    int* out       = (int*)d_out;
    int* qsort_out = out;                        // first  BH*NQ ints
    int* pick_out  = out + (size_t)BH_TOT * NQ;  // second BH*NQ ints

    const size_t smem1 = (size_t)QT * NK * sizeof(float) + (size_t)QT * KSEL * sizeof(int);
    qnf_score_topk_kernel<<<BH_TOT * (NQ / QT), 512, smem1, stream>>>(query, key, sampled);
    sort_pick_kernel<<<BH_TOT, 1024, 0, stream>>>(sampled, qsort_out, pick_out);
}
